// Attention_86998857548403
// MI455X (gfx1250) — matrix operand order repaired and checked by reading, not compiled, not run on hardware
//
#include <hip/hip_runtime.h>
#include <hip/hip_bf16.h>
#include <cstdint>

typedef __bf16 bf16_t;
typedef __attribute__((ext_vector_type(4)))  __bf16 v4bf;
typedef __attribute__((ext_vector_type(8)))  __bf16 v8bf;
typedef __attribute__((ext_vector_type(16))) __bf16 v16bf;
typedef __attribute__((ext_vector_type(8)))  float  v8f;

#define N_TOK 4096
#define DMODEL 1024
#define DFFN   4096
#define EPSV   1e-5f

// ---------------------------------------------------------------- conversions
__global__ void k_f32_to_bf16(const float* __restrict__ in, bf16_t* __restrict__ out, int n) {
  int i = (blockIdx.x * 256 + threadIdx.x) * 4;
  if (i < n) {
    float4 f = *(const float4*)(in + i);
    v4bf o; o[0] = (bf16_t)f.x; o[1] = (bf16_t)f.y; o[2] = (bf16_t)f.z; o[3] = (bf16_t)f.w;
    *(v4bf*)(out + i) = o;
  }
}

// ---------------------------------------------------------------- WMMA GEMM
// out = A[M,K] @ W[Ncols,K]^T (+ bias), M,Ncols multiples of 128; K multiple of 32
// MODE 0: bf16 out                   (q, k)
// MODE 1: bf16 out, transposed +bias (vT)
// MODE 2: f32 out                    (S, attn@v)
// MODE 3: bf16 out + bias            (ffn hidden)
// MODE 4: f32 out + bias             (ffn out)
constexpr int BM = 128, BN = 128, BK = 32;
constexpr int LDSS = BK + 8;   // bf16 row stride: 80 B, keeps 16B alignment, spreads banks

__device__ __forceinline__ void async_b128(uint32_t lds_addr, const void* gaddr) {
  // global -> LDS direct copy, 16B per lane, tracked by ASYNCcnt (no VGPR data path)
  asm volatile("global_load_async_to_lds_b128 %0, %1, off"
               :: "v"(lds_addr), "v"((uint64_t)(uintptr_t)gaddr) : "memory");
}
__device__ __forceinline__ void wait_async0() {
  asm volatile("s_wait_asynccnt 0" ::: "memory");
}

template <int MODE>
__global__ __launch_bounds__(256) void k_gemm_bf16_wmma(
    const bf16_t* __restrict__ A, int lda,
    const bf16_t* __restrict__ W, int ldw,
    const float* __restrict__ bias,
    float* __restrict__ outF, bf16_t* __restrict__ outB,
    int M, int Ncols, int K, int ldT)
{
  __shared__ bf16_t lA[2][BM * LDSS];
  __shared__ bf16_t lB[2][BN * LDSS];

  const int tid  = threadIdx.x;
  const int lane = tid & 31;
  const int wave = tid >> 5;
  const int wm = wave & 1, wn = wave >> 1;        // 2 x 4 wave grid, 64x32 per wave
  const int half = lane >> 4, l16 = lane & 15;
  const int bm0 = blockIdx.y * BM, bn0 = blockIdx.x * BN;

  v8f acc[4][2];
  for (int i = 0; i < 4; ++i)
    for (int j = 0; j < 2; ++j)
      for (int e = 0; e < 8; ++e) acc[i][j][e] = 0.0f;

  // staging: 256 threads x (2 rows) x 16B for each of A,B per tile
  const int ldr = tid >> 2;          // 0..63
  const int ldc = (tid & 3) * 8;     // 0,8,16,24 (bf16), 16B chunks

  const bf16_t* gA0 = &A[(size_t)(bm0 + ldr) * lda + ldc];
  const bf16_t* gA1 = &A[(size_t)(bm0 + 64 + ldr) * lda + ldc];
  const bf16_t* gB0 = &W[(size_t)(bn0 + ldr) * ldw + ldc];
  const bf16_t* gB1 = &W[(size_t)(bn0 + 64 + ldr) * ldw + ldc];

  const uint32_t la0[2] = { (uint32_t)(uintptr_t)&lA[0][ldr * LDSS + ldc],
                            (uint32_t)(uintptr_t)&lA[1][ldr * LDSS + ldc] };
  const uint32_t la1[2] = { (uint32_t)(uintptr_t)&lA[0][(64 + ldr) * LDSS + ldc],
                            (uint32_t)(uintptr_t)&lA[1][(64 + ldr) * LDSS + ldc] };
  const uint32_t lb0[2] = { (uint32_t)(uintptr_t)&lB[0][ldr * LDSS + ldc],
                            (uint32_t)(uintptr_t)&lB[1][ldr * LDSS + ldc] };
  const uint32_t lb1[2] = { (uint32_t)(uintptr_t)&lB[0][(64 + ldr) * LDSS + ldc],
                            (uint32_t)(uintptr_t)&lB[1][(64 + ldr) * LDSS + ldc] };

  auto stage = [&](int buf, int k0) {
    async_b128(la0[buf], gA0 + k0);
    async_b128(la1[buf], gA1 + k0);
    async_b128(lb0[buf], gB0 + k0);
    async_b128(lb1[buf], gB1 + k0);
  };

  const int nk = K / BK;
  stage(0, 0);

  for (int kt = 0; kt < nk; ++kt) {
    const int cur = kt & 1;
    wait_async0();        // this wave's tile kt has landed in LDS
    __syncthreads();      // all waves' tile kt landed; previous buffer reads finished
    if (kt + 1 < nk) stage(1 - cur, (kt + 1) * BK);  // overlap next transfer with compute

    v16bf af[4], bfr[2];
#pragma unroll
    for (int i = 0; i < 4; ++i) {
      // A 16x32 bf16 fragment: lane half 0 -> K{0-7,16-23}, half 1 -> K{8-15,24-31}
      const int row = wm * 64 + i * 16 + l16;
      v8bf lo = *(const v8bf*)&lA[cur][row * LDSS + half * 8];
      v8bf hi = *(const v8bf*)&lA[cur][row * LDSS + 16 + half * 8];
      af[i] = __builtin_shufflevector(lo, hi, 0,1,2,3,4,5,6,7,8,9,10,11,12,13,14,15);
    }
#pragma unroll
    for (int j = 0; j < 2; ++j) {
      // B 32x16 bf16 fragment: lane = column n, half 0 -> K{0-7,16-23}, half 1 -> K{8-15,24-31}
      const int row = wn * 32 + j * 16 + l16;
      v8bf blo = *(const v8bf*)&lB[cur][row * LDSS + half * 8];
      v8bf bhi = *(const v8bf*)&lB[cur][row * LDSS + 16 + half * 8];
      bfr[j] = __builtin_shufflevector(blo, bhi, 0,1,2,3,4,5,6,7,8,9,10,11,12,13,14,15);
    }
#pragma unroll
    for (int i = 0; i < 4; ++i)
#pragma unroll
      for (int j = 0; j < 2; ++j)
        acc[i][j] = __builtin_amdgcn_wmma_f32_16x16x32_bf16(
            false, af[i], false, bfr[j], (short)0, acc[i][j], false, false);
    __syncthreads();      // done reading lA/lB[cur]; it may be restaged next iter
  }

  // epilogue: C/D layout -> element (M = v + 8*half, N = l16)
#pragma unroll
  for (int i = 0; i < 4; ++i) {
#pragma unroll
    for (int j = 0; j < 2; ++j) {
      const int col = bn0 + wn * 32 + j * 16 + l16;
      const float bvv = (MODE == 1 || MODE == 3 || MODE == 4) ? bias[col] : 0.0f;
#pragma unroll
      for (int v = 0; v < 8; ++v) {
        const int row = bm0 + wm * 64 + i * 16 + half * 8 + v;
        const float val = acc[i][j][v] + bvv;
        if (MODE == 0 || MODE == 3)      outB[(size_t)row * Ncols + col] = (bf16_t)val;
        else if (MODE == 1)              outB[(size_t)col * ldT + row]  = (bf16_t)val;
        else                             outF[(size_t)row * Ncols + col] = val;
      }
    }
  }
}

// ---------------------------------------------------------------- softmax
// one block per row; reads fp32 scores, writes bf16 probs (P strided inside S buffer)
__global__ __launch_bounds__(256) void k_softmax_bf16(
    const float* __restrict__ S, bf16_t* __restrict__ P,
    int n, int ldp, float scale)
{
  const int row = blockIdx.x, tid = threadIdx.x;
  const float* s = S + (size_t)row * n;
  bf16_t* p = P + (size_t)row * ldp;
  __shared__ float red[8];

  float vals[16];
  float m = -3.4e38f;
#pragma unroll
  for (int i = 0; i < 16; ++i) { vals[i] = s[tid + i * 256] * scale; m = fmaxf(m, vals[i]); }
#pragma unroll
  for (int off = 16; off > 0; off >>= 1) m = fmaxf(m, __shfl_xor(m, off, 32));
  if ((tid & 31) == 0) red[tid >> 5] = m;
  __syncthreads();
  float rm = red[0];
#pragma unroll
  for (int i = 1; i < 8; ++i) rm = fmaxf(rm, red[i]);
  __syncthreads();

  float sum = 0.0f;
#pragma unroll
  for (int i = 0; i < 16; ++i) { vals[i] = __expf(vals[i] - rm); sum += vals[i]; }
#pragma unroll
  for (int off = 16; off > 0; off >>= 1) sum += __shfl_xor(sum, off, 32);
  if ((tid & 31) == 0) red[tid >> 5] = sum;
  __syncthreads();
  float tot = 0.0f;
#pragma unroll
  for (int i = 0; i < 8; ++i) tot += red[i];
  const float inv = 1.0f / tot;
#pragma unroll
  for (int i = 0; i < 16; ++i) p[tid + i * 256] = (bf16_t)(vals[i] * inv);
}

// ---------------------------------------------------------------- residual + layernorm
__global__ __launch_bounds__(256) void k_add_ln(
    const float* __restrict__ a, const float* __restrict__ b,
    const float* __restrict__ gamma, const float* __restrict__ beta,
    float* __restrict__ outF, bf16_t* __restrict__ outB, int d)
{
  const int row = blockIdx.x, tid = threadIdx.x;
  __shared__ float red[16];
  float v[4], s = 0.0f, ss = 0.0f;
#pragma unroll
  for (int i = 0; i < 4; ++i) {
    const int idx = tid + i * 256;
    const float t = a[(size_t)row * d + idx] + b[(size_t)row * d + idx];
    v[i] = t; s += t; ss += t * t;
  }
#pragma unroll
  for (int off = 16; off > 0; off >>= 1) { s += __shfl_xor(s, off, 32); ss += __shfl_xor(ss, off, 32); }
  if ((tid & 31) == 0) { red[tid >> 5] = s; red[8 + (tid >> 5)] = ss; }
  __syncthreads();
  float S = 0.0f, SS = 0.0f;
#pragma unroll
  for (int i = 0; i < 8; ++i) { S += red[i]; SS += red[8 + i]; }
  const float mu = S / d;
  const float var = SS / d - mu * mu;
  const float rstd = rsqrtf(var + EPSV);
#pragma unroll
  for (int i = 0; i < 4; ++i) {
    const int idx = tid + i * 256;
    const float o = (v[i] - mu) * rstd * gamma[idx] + beta[idx];
    outF[(size_t)row * d + idx] = o;
    if (outB) outB[(size_t)row * d + idx] = (bf16_t)o;
  }
}

// ---------------------------------------------------------------- launcher
extern "C" void kernel_launch(void* const* d_in, const int* in_sizes, int n_in,
                              void* d_out, int out_size, void* d_ws, size_t ws_size,
                              hipStream_t stream) {
  const float* x   = (const float*)d_in[0];
  const float* Wq  = (const float*)d_in[1];
  const float* Wk  = (const float*)d_in[2];
  const float* Wv  = (const float*)d_in[3];
  const float* bv  = (const float*)d_in[4];
  const float* g0  = (const float*)d_in[5];
  const float* b0  = (const float*)d_in[6];
  const float* W1  = (const float*)d_in[7];
  const float* b1  = (const float*)d_in[8];
  const float* W2  = (const float*)d_in[9];
  const float* b2  = (const float*)d_in[10];
  const float* g1  = (const float*)d_in[11];
  const float* b1n = (const float*)d_in[12];

  const int N = N_TOK, D = DMODEL, H = DFFN;

  char* p = (char*)d_ws;
  auto alloc = [&](size_t bytes) { void* q = (void*)p; p += (bytes + 255) & ~(size_t)255; return q; };

  bf16_t* xb   = (bf16_t*)alloc((size_t)N * D * 2);
  bf16_t* wqb  = (bf16_t*)alloc((size_t)D * D * 2);
  bf16_t* wkb  = (bf16_t*)alloc((size_t)D * D * 2);
  bf16_t* wvb  = (bf16_t*)alloc((size_t)D * D * 2);
  bf16_t* w1b  = (bf16_t*)alloc((size_t)H * D * 2);
  bf16_t* w2b  = (bf16_t*)alloc((size_t)D * H * 2);
  bf16_t* qb   = (bf16_t*)alloc((size_t)N * D * 2);
  bf16_t* kb   = (bf16_t*)alloc((size_t)N * D * 2);
  bf16_t* vTb  = (bf16_t*)alloc((size_t)D * N * 2);   // v transposed [D, N]
  float*  R64  = (float*)alloc((size_t)N * N * 4);    // S, then P in-place, then FFN scratch
  float*  attnO= (float*)alloc((size_t)N * D * 4);
  float*  h    = (float*)alloc((size_t)N * D * 4);
  bf16_t* hb   = (bf16_t*)alloc((size_t)N * D * 2);

  float*  Sbuf = R64;
  bf16_t* Pb   = (bf16_t*)R64;                        // row r at bf16 offset r*2N (in place)
  bf16_t* t1b  = (bf16_t*)R64;                        // FFN hidden, reuses S after attention
  float*  ffd  = (float*)((char*)R64 + (size_t)N * H * 2);

  // 1) fp32 -> bf16 casts (bandwidth-trivial)
  k_f32_to_bf16<<<(N * D / 4 + 255) / 256, 256, 0, stream>>>(x,  xb,  N * D);
  k_f32_to_bf16<<<(D * D / 4 + 255) / 256, 256, 0, stream>>>(Wq, wqb, D * D);
  k_f32_to_bf16<<<(D * D / 4 + 255) / 256, 256, 0, stream>>>(Wk, wkb, D * D);
  k_f32_to_bf16<<<(D * D / 4 + 255) / 256, 256, 0, stream>>>(Wv, wvb, D * D);
  k_f32_to_bf16<<<(H * D / 4 + 255) / 256, 256, 0, stream>>>(W1, w1b, H * D);
  k_f32_to_bf16<<<(D * H / 4 + 255) / 256, 256, 0, stream>>>(W2, w2b, D * H);

  // 2) q = x Wq^T, k = x Wk^T  (bf16 out)
  k_gemm_bf16_wmma<0><<<dim3(D / BN, N / BM), 256, 0, stream>>>(
      xb, D, wqb, D, nullptr, nullptr, qb, N, D, D, 0);
  k_gemm_bf16_wmma<0><<<dim3(D / BN, N / BM), 256, 0, stream>>>(
      xb, D, wkb, D, nullptr, nullptr, kb, N, D, D, 0);
  // 3) vT = (x Wv^T + bv)^T  (bf16 transposed out)
  k_gemm_bf16_wmma<1><<<dim3(D / BN, N / BM), 256, 0, stream>>>(
      xb, D, wvb, D, bv, nullptr, vTb, N, D, D, N);

  // 4) S = q k^T  (f32)
  k_gemm_bf16_wmma<2><<<dim3(N / BN, N / BM), 256, 0, stream>>>(
      qb, D, kb, D, nullptr, Sbuf, nullptr, N, N, D, 0);

  // 5) P = softmax(scale*S), bf16, in place (row stride 2N bf16 elements)
  k_softmax_bf16<<<N, 256, 0, stream>>>(Sbuf, Pb, N, 2 * N, 0.03125f /* D^-0.5 */);

  // 6) attnO = P v = P (vT)^T  (f32)
  k_gemm_bf16_wmma<2><<<dim3(D / BN, N / BM), 256, 0, stream>>>(
      Pb, 2 * N, vTb, N, nullptr, attnO, nullptr, N, D, N, 0);

  // 7) h = layernorm(x + attnO), f32 + bf16 copy
  k_add_ln<<<N, 256, 0, stream>>>(x, attnO, g0, b0, h, hb, D);

  // 8) t1 = h W1^T + b1  (bf16, reuses S region)
  k_gemm_bf16_wmma<3><<<dim3(H / BN, N / BM), 256, 0, stream>>>(
      hb, D, w1b, D, b1, nullptr, t1b, N, H, D, 0);

  // 9) ffd = t1 W2^T + b2  (f32)
  k_gemm_bf16_wmma<4><<<dim3(D / BN, N / BM), 256, 0, stream>>>(
      t1b, H, w2b, H, b2, ffd, nullptr, N, D, H, 0);

  // 10) out = layernorm(ffd + h)
  k_add_ln<<<N, 256, 0, stream>>>(ffd, h, g1, b1n, (float*)d_out, nullptr, D);
}
